// MGRU_RBF_18992345383062
// MI455X (gfx1250) — compile-verified, matrix operand order checked by reading, not run on hardware
//
#include <hip/hip_runtime.h>
#include <hip/hip_bf16.h>

typedef __attribute__((ext_vector_type(16))) _Float16 v16h;
typedef __attribute__((ext_vector_type(8)))  _Float16 v8h;
typedef __attribute__((ext_vector_type(8)))  float    v8f;

#define BQ 1024
#define TQ 100
#define FQ 128
#define HQ 512

__device__ __forceinline__ float fast_sigmoid(float x) {
    // 1/(1+e^-x) with v_exp_f32 + v_rcp_f32 (no IEEE divide expansion)
    return __builtin_amdgcn_rcpf(1.0f + __expf(-x));
}
__device__ __forceinline__ float fast_tanh(float x) {
    float e = __expf(2.0f * x);
    return (e - 1.0f) * __builtin_amdgcn_rcpf(e + 1.0f);
}

// ---------------------------------------------------------------------------
// Utility kernels
// ---------------------------------------------------------------------------
__global__ void mgru_zero_f32(float* __restrict__ p, int n) {
    int i = blockIdx.x * blockDim.x + threadIdx.x;
    if (i < n) p[i] = 0.0f;
}

// Convert weight W[N,K] (f32) -> f16 same layout. zero_diag masks W[n,n].
__global__ void mgru_conv_w16(const float* __restrict__ W, _Float16* __restrict__ W16,
                              int N, int K, int zero_diag) {
    int idx = blockIdx.x * blockDim.x + threadIdx.x;
    if (idx >= N * K) return;
    int n = idx / K, k = idx - n * K;
    float w = W[idx];
    if (zero_diag && n == k) w = 0.0f;
    W16[idx] = (_Float16)w;
}

// ---------------------------------------------------------------------------
// WMMA GEMM: C[M,N] = A[M,K] (f32, row stride lda) * W16[N,K]^T (f16) + bias[N]
// mode 0: plain   mode 1: sigmoid   mode 2: h_dec = extra * exp(-relu(x))
// Block: 128 threads (4 waves, 2x2), 64x64 tile, K step 32.
// Both A and B tiles staged in LDS with per-lane-contiguous halves so every
// fragment is two ds_load_b128.
// ---------------------------------------------------------------------------
__global__ __launch_bounds__(128) void mgru_gemm_wmma(
    const float* __restrict__ A, long lda,
    const _Float16* __restrict__ W16,   // [N,K] row-major f16
    const float* __restrict__ bias,
    float* __restrict__ C, int N, int K,
    int mode, const float* __restrict__ extra) {

    __shared__ _Float16 As[64][40];   // [m][k], 16B-aligned rows (pitch 80B)
    __shared__ _Float16 Bs[64][40];   // [n][k]

    const int tid  = threadIdx.x;
    const int lane = tid & 31;
    const int wave = tid >> 5;
    const int wrow = wave >> 1;       // 0..1
    const int wcol = wave & 1;        // 0..1
    const int m0 = blockIdx.y * 64;
    const int n0 = blockIdx.x * 64;

    v8f acc[2][2];
    #pragma unroll
    for (int i = 0; i < 2; ++i)
        #pragma unroll
        for (int j = 0; j < 2; ++j) {
            v8f z = {};
            acc[i][j] = z;
        }

    const int l16 = lane & 15;
    const int kb  = (lane >> 4) * 8;  // 0 or 8

    const int r  = tid >> 1;          // 0..63 (tile row for A, tile col for B)
    const int kh = (tid & 1) * 16;    // 0 or 16

    for (int k0 = 0; k0 < K; k0 += 32) {
        // ---- Stage A tile 64x32 (f32 -> f16), vector loads/stores ----
        {
            const float4* ap =
                reinterpret_cast<const float4*>(A + (long)(m0 + r) * lda + (k0 + kh));
            float4 f0 = ap[0], f1 = ap[1], f2 = ap[2], f3 = ap[3];
            if (k0 + 32 < K) __builtin_prefetch(ap + 8, 0, 1);
            v8h h0 = {(_Float16)f0.x, (_Float16)f0.y, (_Float16)f0.z, (_Float16)f0.w,
                      (_Float16)f1.x, (_Float16)f1.y, (_Float16)f1.z, (_Float16)f1.w};
            v8h h1 = {(_Float16)f2.x, (_Float16)f2.y, (_Float16)f2.z, (_Float16)f2.w,
                      (_Float16)f3.x, (_Float16)f3.y, (_Float16)f3.z, (_Float16)f3.w};
            *reinterpret_cast<v8h*>(&As[r][kh])     = h0;
            *reinterpret_cast<v8h*>(&As[r][kh + 8]) = h1;
        }
        // ---- Stage B tile 64(cols)x32(k) from W16[N,K] (f16, contiguous k) ----
        {
            const v8h* bp =
                reinterpret_cast<const v8h*>(W16 + (long)(n0 + r) * K + (k0 + kh));
            v8h b0 = bp[0], b1 = bp[1];
            if (k0 + 32 < K) __builtin_prefetch(bp + 4, 0, 1);
            *reinterpret_cast<v8h*>(&Bs[r][kh])     = b0;
            *reinterpret_cast<v8h*>(&Bs[r][kh + 8]) = b1;
        }
        __syncthreads();

        // ---- Fragments: two ds_load_b128 per operand, then 4 WMMAs ----
        v16h afrag[2], bfrag[2];
        #pragma unroll
        for (int mf = 0; mf < 2; ++mf) {
            int row = wrow * 32 + mf * 16 + l16;
            v8h lo = *reinterpret_cast<const v8h*>(&As[row][kb]);
            v8h hi = *reinterpret_cast<const v8h*>(&As[row][16 + kb]);
            #pragma unroll
            for (int j = 0; j < 8; ++j) { afrag[mf][j] = lo[j]; afrag[mf][j + 8] = hi[j]; }
        }
        #pragma unroll
        for (int nf = 0; nf < 2; ++nf) {
            int col = wcol * 32 + nf * 16 + l16;
            v8h lo = *reinterpret_cast<const v8h*>(&Bs[col][kb]);
            v8h hi = *reinterpret_cast<const v8h*>(&Bs[col][16 + kb]);
            #pragma unroll
            for (int j = 0; j < 8; ++j) { bfrag[nf][j] = lo[j]; bfrag[nf][j + 8] = hi[j]; }
        }
        #pragma unroll
        for (int mf = 0; mf < 2; ++mf)
            #pragma unroll
            for (int nf = 0; nf < 2; ++nf)
                acc[mf][nf] = __builtin_amdgcn_wmma_f32_16x16x32_f16(
                    false, afrag[mf], false, bfrag[nf],
                    (short)0, acc[mf][nf], false, false);
        __syncthreads();
    }

    // Epilogue (C layout: VGPR i -> M = i + 8*(lane>=16), N = lane&15).
    const int hi8 = (lane >> 4) * 8;
    #pragma unroll
    for (int mf = 0; mf < 2; ++mf) {
        #pragma unroll
        for (int nf = 0; nf < 2; ++nf) {
            int colg = n0 + wcol * 32 + nf * 16 + l16;
            float bv = bias[colg];
            #pragma unroll
            for (int i = 0; i < 8; ++i) {
                int rowg = m0 + wrow * 32 + mf * 16 + hi8 + i;
                float x = acc[mf][nf][i] + bv;
                long o = (long)rowg * N + colg;
                if (mode == 0) {
                    C[o] = x;
                } else if (mode == 1) {
                    C[o] = fast_sigmoid(x);
                } else { // mode 2: gamma_h decay fused with hidden multiply
                    float g = __expf(-fmaxf(x, 0.0f));
                    C[o] = extra[o] * g;
                }
            }
        }
    }
}

// ---------------------------------------------------------------------------
// Elementwise stage after x_hat GEMM: gamma_x, x_c, cat1=[x_c,r], cat2=[gx,m],
// MAE numerator term0 and mask denominator.
// ---------------------------------------------------------------------------
__global__ __launch_bounds__(256) void mgru_ew_pre(
    const float* __restrict__ inp, const float* __restrict__ w_dx,
    const float* __restrict__ b_dx, const float* __restrict__ x_hat,
    float* __restrict__ cat1, float* __restrict__ cat2,
    float* __restrict__ num, float* __restrict__ den, int t) {
    int idx = blockIdx.x * 256 + threadIdx.x;
    float ln = 0.0f, lm = 0.0f;
    if (idx < BQ * FQ) {
        int b = idx >> 7, f = idx & (FQ - 1);
        size_t ib = ((size_t)b * 4 * TQ + t) * FQ + f;
        float v  = inp[ib];
        float d  = inp[ib + (size_t)TQ * FQ];
        float m  = inp[ib + 2 * (size_t)TQ * FQ];
        float rb = inp[ib + 3 * (size_t)TQ * FQ];
        float gx = __expf(-fmaxf(d * w_dx[f] + b_dx[f], 0.0f));
        float xh = x_hat[idx];
        float xc = m * v + (1.0f - m) * xh;
        cat1[(size_t)b * 2 * FQ + f]      = xc;
        cat1[(size_t)b * 2 * FQ + FQ + f] = rb;
        cat2[(size_t)b * 2 * FQ + f]      = gx;
        cat2[(size_t)b * 2 * FQ + FQ + f] = m;
        ln = fabsf(v - xh) * m;
        lm = m;
    }
    __shared__ float s0[256], s1[256];
    s0[threadIdx.x] = ln; s1[threadIdx.x] = lm;
    __syncthreads();
    for (int s = 128; s > 0; s >>= 1) {
        if (threadIdx.x < s) { s0[threadIdx.x] += s0[threadIdx.x + s];
                               s1[threadIdx.x] += s1[threadIdx.x + s]; }
        __syncthreads();
    }
    if (threadIdx.x == 0) {
        atomicAdd(&num[t * 4 + 0], s0[0]);
        atomicAdd(&den[t], s1[0]);
    }
}

// After concat_hat GEMM: a_c, MAE term1.
__global__ __launch_bounds__(256) void mgru_ew_cat(
    const float* __restrict__ inp, const float* __restrict__ concat_hat,
    float* __restrict__ a_c, float* __restrict__ num, int t) {
    int idx = blockIdx.x * 256 + threadIdx.x;
    float ln = 0.0f;
    if (idx < BQ * FQ) {
        int b = idx >> 7, f = idx & (FQ - 1);
        size_t ib = ((size_t)b * 4 * TQ + t) * FQ + f;
        float v = inp[ib];
        float m = inp[ib + 2 * (size_t)TQ * FQ];
        float ch = concat_hat[idx];
        a_c[idx] = m * v + (1.0f - m) * ch;
        ln = fabsf(v - ch) * m;
    }
    __shared__ float s0[256];
    s0[threadIdx.x] = ln;
    __syncthreads();
    for (int s = 128; s > 0; s >>= 1) {
        if (threadIdx.x < s) s0[threadIdx.x] += s0[threadIdx.x + s];
        __syncthreads();
    }
    if (threadIdx.x == 0) atomicAdd(&num[t * 4 + 1], s0[0]);
}

// After z_hat + beta: c_hat, c_c, outputs, cat3=[c_c,m], MAE terms 2 & 3.
__global__ __launch_bounds__(256) void mgru_ew_chat(
    const float* __restrict__ inp, const float* __restrict__ z_hat,
    const float* __restrict__ beta, const float* __restrict__ x_hat,
    float* __restrict__ cat3, float* __restrict__ out_chat,
    float* __restrict__ out_imp, float* __restrict__ num, int t) {
    int idx = blockIdx.x * 256 + threadIdx.x;
    float l2 = 0.0f, l3 = 0.0f;
    if (idx < BQ * FQ) {
        int b = idx >> 7, f = idx & (FQ - 1);
        size_t ib = ((size_t)b * 4 * TQ + t) * FQ + f;
        float v = inp[ib];
        float m = inp[ib + 2 * (size_t)TQ * FQ];
        float zh = z_hat[idx];
        float be = beta[idx];
        float xh = x_hat[idx];
        float chat = be * zh + (1.0f - be) * xh;
        float cc = m * v + (1.0f - m) * chat;
        size_t oo = ((size_t)b * TQ + t) * FQ + f;
        out_chat[oo] = chat;
        out_imp[oo]  = cc;
        cat3[(size_t)b * 2 * FQ + f]      = cc;
        cat3[(size_t)b * 2 * FQ + FQ + f] = m;
        l2 = fabsf(v - zh) * m;
        l3 = fabsf(v - chat) * m;
    }
    __shared__ float s0[256], s1[256];
    s0[threadIdx.x] = l2; s1[threadIdx.x] = l3;
    __syncthreads();
    for (int s = 128; s > 0; s >>= 1) {
        if (threadIdx.x < s) { s0[threadIdx.x] += s0[threadIdx.x + s];
                               s1[threadIdx.x] += s1[threadIdx.x + s]; }
        __syncthreads();
    }
    if (threadIdx.x == 0) {
        atomicAdd(&num[t * 4 + 2], s0[0]);
        atomicAdd(&num[t * 4 + 3], s1[0]);
    }
}

// GRU gate fusion: hid <- (1-z)*tanh(i_n + r*h_n) + z*h_dec
__global__ __launch_bounds__(256) void mgru_gru_fuse(
    const float* __restrict__ gi, const float* __restrict__ gh,
    const float* __restrict__ h_dec, float* __restrict__ hid) {
    int idx = blockIdx.x * 256 + threadIdx.x;
    if (idx >= BQ * HQ) return;
    int b = idx >> 9, h = idx & (HQ - 1);
    size_t base = (size_t)b * 3 * HQ + h;
    float ir = gi[base], iz = gi[base + HQ], in_ = gi[base + 2 * HQ];
    float hr = gh[base], hz = gh[base + HQ], hn  = gh[base + 2 * HQ];
    float rg = fast_sigmoid(ir + hr);
    float zg = fast_sigmoid(iz + hz);
    float ng = fast_tanh(in_ + rg * hn);
    float hd = h_dec[idx];
    hid[idx] = (1.0f - zg) * ng + zg * hd;
}

// Final loss: x_loss = 5 * sum_t (2*n0 + n1 + n2 + n3) / (den_t + 1e-5)
__global__ __launch_bounds__(128) void mgru_loss_final(
    const float* __restrict__ num, const float* __restrict__ den,
    float* __restrict__ out) {
    __shared__ float s0[128];
    float s = 0.0f;
    for (int t = threadIdx.x; t < TQ; t += 128) {
        float dn = den[t] + 1e-5f;
        s += (2.0f * num[t * 4] + num[t * 4 + 1] + num[t * 4 + 2] + num[t * 4 + 3]) / dn;
    }
    s0[threadIdx.x] = s;
    __syncthreads();
    for (int st = 64; st > 0; st >>= 1) {
        if (threadIdx.x < st) s0[threadIdx.x] += s0[threadIdx.x + st];
        __syncthreads();
    }
    if (threadIdx.x == 0) out[0] = 5.0f * s0[0];
}

// ---------------------------------------------------------------------------
// Launch
// ---------------------------------------------------------------------------
extern "C" void kernel_launch(void* const* d_in, const int* in_sizes, int n_in,
                              void* d_out, int out_size, void* d_ws, size_t ws_size,
                              hipStream_t stream) {
    (void)in_sizes; (void)n_in; (void)out_size; (void)ws_size;

    const float* inp    = (const float*)d_in[0];
    const float* W_dh   = (const float*)d_in[1];
    const float* b_dh   = (const float*)d_in[2];
    const float* w_dx   = (const float*)d_in[3];
    const float* b_dx   = (const float*)d_in[4];
    const float* W_out  = (const float*)d_in[5];
    const float* b_out  = (const float*)d_in[6];
    const float* W_z    = (const float*)d_in[7];
    const float* b_z    = (const float*)d_in[8];
    const float* W_cat  = (const float*)d_in[9];
    const float* b_cat  = (const float*)d_in[10];
    const float* W_beta = (const float*)d_in[11];
    const float* b_beta = (const float*)d_in[12];
    const float* W_ih   = (const float*)d_in[13];
    const float* b_ih   = (const float*)d_in[14];
    const float* W_hh   = (const float*)d_in[15];
    const float* b_hh   = (const float*)d_in[16];

    float* out_chat = (float*)d_out;
    float* out_imp  = out_chat + (size_t)BQ * TQ * FQ;
    float* out_loss = out_chat + 2 * (size_t)BQ * TQ * FQ;

    // Workspace layout
    char* ws = (char*)d_ws;
    size_t off = 0;
    auto alloc = [&](size_t bytes) -> char* {
        char* p = ws + off;
        off += (bytes + 255) & ~(size_t)255;
        return p;
    };
    _Float16* W16_dh   = (_Float16*)alloc((size_t)HQ * FQ * 2);
    _Float16* W16_out  = (_Float16*)alloc((size_t)FQ * HQ * 2);
    _Float16* W16_z    = (_Float16*)alloc((size_t)FQ * FQ * 2);
    _Float16* W16_cat  = (_Float16*)alloc((size_t)FQ * 2 * FQ * 2);
    _Float16* W16_beta = (_Float16*)alloc((size_t)FQ * 2 * FQ * 2);
    _Float16* W16_ih   = (_Float16*)alloc((size_t)3 * HQ * 2 * FQ * 2);
    _Float16* W16_hh   = (_Float16*)alloc((size_t)3 * HQ * HQ * 2);

    float* hid    = (float*)alloc((size_t)BQ * HQ * 4);
    float* h_dec  = (float*)alloc((size_t)BQ * HQ * 4);
    float* x_hat  = (float*)alloc((size_t)BQ * FQ * 4);
    float* cat1   = (float*)alloc((size_t)BQ * 2 * FQ * 4);
    float* cat2   = (float*)alloc((size_t)BQ * 2 * FQ * 4);
    float* cat3   = (float*)alloc((size_t)BQ * 2 * FQ * 4);
    float* c_hatb = (float*)alloc((size_t)BQ * FQ * 4);  // concat_hat
    float* a_c    = (float*)alloc((size_t)BQ * FQ * 4);
    float* z_hat  = (float*)alloc((size_t)BQ * FQ * 4);
    float* betab  = (float*)alloc((size_t)BQ * FQ * 4);
    float* gi     = (float*)alloc((size_t)BQ * 3 * HQ * 4);
    float* gh     = (float*)alloc((size_t)BQ * 3 * HQ * 4);
    float* numb   = (float*)alloc((size_t)4 * TQ * 4);
    float* denb   = (float*)alloc((size_t)TQ * 4);

    // Init state / accumulators
    mgru_zero_f32<<<(BQ * HQ + 255) / 256, 256, 0, stream>>>(hid, BQ * HQ);
    mgru_zero_f32<<<(4 * TQ + 255) / 256, 256, 0, stream>>>(numb, 4 * TQ);
    mgru_zero_f32<<<(TQ + 255) / 256, 256, 0, stream>>>(denb, TQ);

    // Weight conversions (f32 [N,K] -> f16 [N,K])
    auto conv = [&](const float* W, _Float16* W16, int N, int K, int zd) {
        mgru_conv_w16<<<(N * K + 255) / 256, 256, 0, stream>>>(W, W16, N, K, zd);
    };
    conv(W_dh,   W16_dh,   HQ,     FQ,     0);
    conv(W_out,  W16_out,  FQ,     HQ,     0);
    conv(W_z,    W16_z,    FQ,     FQ,     1);   // off-diagonal mask
    conv(W_cat,  W16_cat,  FQ,     2 * FQ, 0);
    conv(W_beta, W16_beta, FQ,     2 * FQ, 0);
    conv(W_ih,   W16_ih,   3 * HQ, 2 * FQ, 0);
    conv(W_hh,   W16_hh,   3 * HQ, HQ,     0);

    auto gemm = [&](const float* A, long lda, const _Float16* W16,
                    const float* bias, float* C, int N, int K,
                    int mode, const float* extra) {
        dim3 g(N / 64, BQ / 64);
        mgru_gemm_wmma<<<g, 128, 0, stream>>>(A, lda, W16, bias, C, N, K, mode, extra);
    };

    const int ewg = (BQ * FQ + 255) / 256;
    const long TF = (long)TQ * FQ;

    for (int t = 0; t < TQ; ++t) {
        // x_hat = hid @ W_out.T + b_out
        gemm(hid, HQ, W16_out, b_out, x_hat, FQ, HQ, 0, nullptr);
        // gamma_x, x_c, cat buffers, loss term 0
        mgru_ew_pre<<<ewg, 256, 0, stream>>>(inp, w_dx, b_dx, x_hat,
                                             cat1, cat2, numb, denb, t);
        // concat_hat = [x_c, r] @ W_cat.T + b_cat
        gemm(cat1, 2 * FQ, W16_cat, b_cat, c_hatb, FQ, 2 * FQ, 0, nullptr);
        // beta = sigmoid([gamma_x, m] @ W_beta.T + b_beta)
        gemm(cat2, 2 * FQ, W16_beta, b_beta, betab, FQ, 2 * FQ, 1, nullptr);
        // a_c, loss term 1
        mgru_ew_cat<<<ewg, 256, 0, stream>>>(inp, c_hatb, a_c, numb, t);
        // z_hat = a_c @ W_z_m.T + b_z
        gemm(a_c, FQ, W16_z, b_z, z_hat, FQ, FQ, 0, nullptr);
        // c_hat, c_c, outputs, cat3, loss terms 2/3
        mgru_ew_chat<<<ewg, 256, 0, stream>>>(inp, z_hat, betab, x_hat,
                                              cat3, out_chat, out_imp, numb, t);
        // h_dec = hid * exp(-relu(delta_t @ W_dh.T + b_dh))  (fused epilogue)
        gemm(inp + TF + (long)t * FQ, 4 * TF, W16_dh, b_dh, h_dec, HQ, FQ, 2, hid);
        // gi = [c_c, m] @ W_ih.T + b_ih
        gemm(cat3, 2 * FQ, W16_ih, b_ih, gi, 3 * HQ, 2 * FQ, 0, nullptr);
        // gh = h_dec @ W_hh.T + b_hh
        gemm(h_dec, HQ, W16_hh, b_hh, gh, 3 * HQ, HQ, 0, nullptr);
        // hid <- GRU(gi, gh, h_dec)
        mgru_gru_fuse<<<(BQ * HQ + 255) / 256, 256, 0, stream>>>(gi, gh, h_dec, hid);
    }

    mgru_loss_final<<<1, 128, 0, stream>>>(numb, denb, out_loss);
}
